// SpatialLanguageModel_49615462204077
// MI455X (gfx1250) — matrix operand order repaired and checked by reading, not compiled, not run on hardware
//
#include <hip/hip_runtime.h>
#include <math.h>

typedef _Float16 h16;
typedef __attribute__((ext_vector_type(8)))  _Float16 v8h;
typedef __attribute__((ext_vector_type(16))) _Float16 v16h;
typedef __attribute__((ext_vector_type(8)))  float    v8f;
typedef __attribute__((ext_vector_type(4)))  unsigned int u32x4;
typedef __attribute__((ext_vector_type(8)))  int i32x8;
typedef __attribute__((ext_vector_type(4)))  int i32x4;

#define B_   4
#define T_   1024
#define V_   50257
#define D_   1024
#define H_   16
#define DH_  64
#define L_   4
#define DFF_ 4096
#define M_   (B_*T_)

#define BM 128
#define BN 64
#define BK 32

#if defined(__HIP_DEVICE_COMPILE__) && \
    __has_builtin(__builtin_amdgcn_tensor_load_to_lds) && \
    __has_builtin(__builtin_amdgcn_s_wait_tensorcnt)
#define USE_TDM 1
#else
#define USE_TDM 0
#endif

static __device__ __forceinline__ v8f wmma32(v16h a, v16h b, v8f c) {
  // D = A(16x32 f16) * B(32x16 f16) + C(16x16 f32)
  return __builtin_amdgcn_wmma_f32_16x16x32_f16(false, a, false, b, (short)0, c,
                                                false, false);
}

static __device__ __forceinline__ v8f zero8f() {
  v8f z = {0.f,0.f,0.f,0.f,0.f,0.f,0.f,0.f};
  return z;
}

#if USE_TDM
// Issue one TDM 2-D tile load (f16 elements) into LDS.
// D# layout per CDNA5 ISA.
// This toolchain exposes the 6-arg builtin:
//   (u32x4 g0, i32x8 g1, i32x4 g2, i32x4 g3, i32x8 extra, i32 cpol)
static __device__ __forceinline__ void tdm_load_tile_f16(
    unsigned lds_off, const void* gptr,
    unsigned tile_w, unsigned tile_h,          // tile dims (elements / rows)
    unsigned tensor_w, unsigned tensor_h,      // OOB clamp dims
    unsigned long long row_stride)             // elements between rows
{
  const unsigned long long ga = (unsigned long long)gptr;
  u32x4 g0;
  g0[0] = 1u;                                           // count=1, is_restore=0, gather=0
  g0[1] = lds_off;                                      // lds_addr (bytes)
  g0[2] = (unsigned)(ga & 0xffffffffull);               // global_addr[31:0]
  g0[3] = (unsigned)((ga >> 32) & 0x01ffffffull) | (2u << 30);  // global_addr[56:32] | type=2
  i32x8 g1;
  g1[0] = (int)(1u << 16);                              // wg_mask=0, data_size=1 (2B)
  g1[1] = (int)((tensor_w & 0xffffu) << 16);            // tensor_dim0[15:0] @ bits63:48
  g1[2] = (int)((tensor_w >> 16) & 0xffffu) |           // tensor_dim0[31:16]
          (int)((tensor_h & 0xffffu) << 16);            // tensor_dim1[15:0]
  g1[3] = (int)((tensor_h >> 16) & 0xffffu) |           // tensor_dim1[31:16]
          (int)((tile_w & 0xffffu) << 16);              // tile_dim0
  g1[4] = (int)(tile_h & 0xffffu);                      // tile_dim1 (tile_dim2=0)
  g1[5] = (int)(row_stride & 0xffffffffull);            // tensor_dim0_stride[31:0]
  g1[6] = (int)((row_stride >> 32) & 0xffffull);        // tensor_dim0_stride[47:32]
  g1[7] = 0;                                            // tensor_dim1_stride hi (unused)
  const i32x4 gz4 = {0, 0, 0, 0};                       // groups 2/3 disabled (<=2D)
  const i32x8 gz8 = {0, 0, 0, 0, 0, 0, 0, 0};
  __builtin_amdgcn_tensor_load_to_lds(g0, g1, gz4, gz4, gz8, 0);
}
#endif

// ---------------------------------------------------------------------------
// Embedding: X[m,:] = tok_emb[idx[m],:] + pos_emb[m%T,:]
// ---------------------------------------------------------------------------
__global__ __launch_bounds__(256) void embed_kernel(
    const int* __restrict__ idx, const float* __restrict__ tok,
    const float* __restrict__ pos, float* __restrict__ X)
{
  const int m = blockIdx.x;
  const int t = m % T_;
  const int token = idx[m];
  const float* te = tok + (size_t)token * D_;
  const float* pe = pos + (size_t)t * D_;
  float* xr = X + (size_t)m * D_;
  for (int d = threadIdx.x; d < D_; d += 256) xr[d] = te[d] + pe[d];
}

// ---------------------------------------------------------------------------
// Row LayerNorm (f32 in) -> f16 out
// ---------------------------------------------------------------------------
__global__ __launch_bounds__(256) void layernorm_f16_kernel(
    const float* __restrict__ X, const float* __restrict__ s,
    const float* __restrict__ b, h16* __restrict__ out, int Dd)
{
  const int row = blockIdx.x;
  const float* x = X + (size_t)row * Dd;
  float sum = 0.f, sq = 0.f;
  for (int d = threadIdx.x; d < Dd; d += 256) { float v = x[d]; sum += v; sq += v*v; }
  #pragma unroll
  for (int o = 16; o; o >>= 1) { sum += __shfl_xor(sum, o, 32); sq += __shfl_xor(sq, o, 32); }
  __shared__ float ssum[8], ssq[8];
  const int wave = threadIdx.x >> 5, lane = threadIdx.x & 31;
  if (lane == 0) { ssum[wave] = sum; ssq[wave] = sq; }
  __syncthreads();
  if (wave == 0) {
    float a = (lane < 8) ? ssum[lane] : 0.f;
    float c = (lane < 8) ? ssq[lane]  : 0.f;
    #pragma unroll
    for (int o = 4; o; o >>= 1) { a += __shfl_xor(a, o, 32); c += __shfl_xor(c, o, 32); }
    if (lane == 0) { ssum[0] = a; ssq[0] = c; }
  }
  __syncthreads();
  const float mu   = ssum[0] / (float)Dd;
  const float var  = ssq[0] / (float)Dd - mu * mu;
  const float rstd = rsqrtf(var + 1e-5f);
  for (int d = threadIdx.x; d < Dd; d += 256)
    out[(size_t)row * Dd + d] = (h16)((x[d] - mu) * rstd * s[d] + b[d]);
}

// ---------------------------------------------------------------------------
// Elementwise helpers
// ---------------------------------------------------------------------------
__global__ __launch_bounds__(256) void f32_to_f16_kernel(
    const float* __restrict__ in, h16* __restrict__ out, size_t n)
{
  size_t i = (size_t)blockIdx.x * 256 + threadIdx.x;
  const size_t stride = (size_t)gridDim.x * 256;
  for (; i < n; i += stride) out[i] = (h16)in[i];
}

__global__ __launch_bounds__(256) void add_inplace_kernel(
    float* __restrict__ x, const float* __restrict__ y, size_t n)
{
  size_t i = (size_t)blockIdx.x * 256 + threadIdx.x;
  const size_t stride = (size_t)gridDim.x * 256;
  for (; i < n; i += stride) x[i] += y[i];
}

__global__ __launch_bounds__(256) void gelu_f16_kernel(
    const float* __restrict__ in, h16* __restrict__ out, size_t n)
{
  size_t i = (size_t)blockIdx.x * 256 + threadIdx.x;
  const size_t stride = (size_t)gridDim.x * 256;
  for (; i < n; i += stride) {
    float v = in[i];
    out[i] = (h16)(0.5f * v * (1.f + erff(v * 0.70710678118f)));
  }
}

// ---------------------------------------------------------------------------
// WMMA GEMM: C[M,N](f32) = A[M,K](f16,row) * W[N,K](f16,row)^T (+ bias[N])
// Block tile 128x64, 8 waves (4x2), each wave 32x32 = 2x2 WMMA tiles, BK=32.
// Tiles staged into LDS by the Tensor Data Mover (TENSORcnt) when available.
// Assumes M % 128 == 0 and K % 32 == 0 (true for all call sites).
// ---------------------------------------------------------------------------
__global__ __launch_bounds__(256) void gemm_f16_kernel(
    const h16* __restrict__ A, const h16* __restrict__ W,
    float* __restrict__ C, const float* __restrict__ bias,
    int M, int N, int K)
{
  __shared__ h16 As[BM][BK];
  __shared__ h16 Bs[BN][BK];
  const int tid  = threadIdx.x;
  const int lane = tid & 31;
  const int wave = tid >> 5;
  const int wr = wave & 3;   // 32-row slab within the 128-row tile
  const int wc = wave >> 2;  // 32-col slab within the 64-col tile
  const int bm = blockIdx.y * BM;
  const int bn = blockIdx.x * BN;

  v8f acc[2][2];
  #pragma unroll
  for (int i = 0; i < 2; ++i)
    #pragma unroll
    for (int j = 0; j < 2; ++j) acc[i][j] = zero8f();

  const int m_lane = lane & 15;
  const int koff   = (lane >> 4) * 16;  // A fragment K sub-offset
  const int kbase  = (lane >> 4) * 16;  // B fragment K sub-offset

#if USE_TDM
  const unsigned as_off = (unsigned)(size_t)(&As[0][0]);
  const unsigned bs_off = (unsigned)(size_t)(&Bs[0][0]);
  const unsigned nb = (unsigned)((N - bn) < BN ? (N - bn) : BN);
#endif

  for (int k0 = 0; k0 < K; k0 += BK) {
#if USE_TDM
    if (wave == 0) {
      // Async DMA of both tiles; zero-fill past tensor_dim (N edge).
      tdm_load_tile_f16(as_off, A + (size_t)bm * K + k0,
                        BK, BM, BK, BM, (unsigned long long)K);
      tdm_load_tile_f16(bs_off, W + (size_t)bn * K + k0,
                        BK, BN, BK, nb, (unsigned long long)K);
      __builtin_amdgcn_s_wait_tensorcnt(0);
    }
    __syncthreads();
#else
    // A tile: 128x32 halves == 1024 8-byte chunks, 4 per thread (no guards:
    // M is a multiple of 128 at every call site).
    #pragma unroll
    for (int i = 0; i < 4; ++i) {
      const int linear = i * 256 + tid;
      const int r  = linear >> 3;
      const int c4 = (linear & 7) * 4;
      *(unsigned long long*)&As[r][c4] =
          *(const unsigned long long*)(A + (size_t)(bm + r) * K + k0 + c4);
    }
    // B tile: 64x32 halves == 512 chunks, 2 per thread; clamp+select for the
    // N edge (v_cndmask instead of exec-mask branches).
    #pragma unroll
    for (int i = 0; i < 2; ++i) {
      const int linear = i * 256 + tid;
      const int r  = linear >> 3;
      const int c4 = (linear & 7) * 4;
      const int gn = bn + r;
      const int gc = gn < N ? gn : (N - 1);
      unsigned long long val =
          *(const unsigned long long*)(W + (size_t)gc * K + k0 + c4);
      *(unsigned long long*)&Bs[r][c4] = (gn < N) ? val : 0ull;
    }
    if (k0 + BK < K) {  // hint next tiles toward the caches
      __builtin_prefetch(A + (size_t)(bm + (tid >> 1)) * K + k0 + BK, 0, 1);
      const int pb = bn + (tid >> 2);
      if (pb < N) __builtin_prefetch(W + (size_t)pb * K + k0 + BK, 0, 1);
    }
    __syncthreads();
#endif

    v16h afrag[2], bfrag[2];
    #pragma unroll
    for (int tm = 0; tm < 2; ++tm) {
      const h16* base = &As[wr * 32 + tm * 16 + m_lane][0];
      v8h lo = *(const v8h*)(base + koff);
      v8h hi = *(const v8h*)(base + koff + 8);
      v16h a;
      #pragma unroll
      for (int e = 0; e < 8; ++e) { a[e] = lo[e]; a[e + 8] = hi[e]; }
      afrag[tm] = a;
    }
    #pragma unroll
    for (int tn = 0; tn < 2; ++tn) {
      const h16* base = &Bs[wc * 32 + tn * 16 + m_lane][0];
      bfrag[tn] = *(const v16h*)(base + kbase);
    }
    #pragma unroll
    for (int tm = 0; tm < 2; ++tm)
      #pragma unroll
      for (int tn = 0; tn < 2; ++tn)
        acc[tm][tn] = wmma32(afrag[tm], bfrag[tn], acc[tm][tn]);
    __syncthreads();
  }

  const int rsel = (lane >> 4) * 8;
  #pragma unroll
  for (int tm = 0; tm < 2; ++tm)
    #pragma unroll
    for (int tn = 0; tn < 2; ++tn) {
      const int rbase = bm + wr * 32 + tm * 16 + rsel;
      const int col   = bn + wc * 32 + tn * 16 + m_lane;
      if (col < N) {
        const float badd = bias ? bias[col] : 0.0f;
        #pragma unroll
        for (int e = 0; e < 8; ++e)
          C[(size_t)(rbase + e) * N + col] = acc[tm][tn][e] + badd;
      }
    }
}

// ---------------------------------------------------------------------------
// Fused flash-style attention. One wave per 16-row query block (4 waves/WG).
// S = Q*K^T via WMMA, online softmax (+ distance & entropy accumulators),
// O += P*V via WMMA (P transposed C-layout -> A-layout through LDS).
// ---------------------------------------------------------------------------
__global__ __launch_bounds__(128) void attention_kernel(
    const float* __restrict__ qkv,     // [M, 3D]
    float* __restrict__ O,             // [M, D]
    const float* __restrict__ dscales, // [H]
    float* __restrict__ acc_defe)      // acc[0]=tot_de, acc[1]=tot_fe
{
  __shared__ h16 Pt[4][16 * 32];
  const int lane = threadIdx.x & 31;
  const int wave = threadIdx.x >> 5;
  const int q0 = (blockIdx.x * 4 + wave) * 16;
  const int h  = blockIdx.y;
  const int b  = blockIdx.z;
  const int brow = b * T_;
  const float dscale = dscales[h];
  const float sscale = 0.125f;  // 1/sqrt(DH)

  const int nl   = lane & 15;
  const int koff = (lane >> 4) * 16;
  const int kb   = (lane >> 4) * 16;
  const int rsel = (lane >> 4) * 8;

  // Q fragments: two K-chunks of 32 covering DH=64
  v16h aq[2];
  {
    const float* qrow = qkv + (size_t)(brow + q0 + nl) * (3 * D_) + h * DH_;
    #pragma unroll
    for (int c = 0; c < 2; ++c) {
      const float* p = qrow + c * 32 + koff;
      v16h a;
      #pragma unroll
      for (int e = 0; e < 8; ++e) { a[e] = (h16)p[e]; a[e + 8] = (h16)p[e + 8]; }
      aq[c] = a;
    }
  }

  float rmax[8], rZ[8], rS1[8], rSd[8];
  #pragma unroll
  for (int e = 0; e < 8; ++e) { rmax[e] = -3.0e38f; rZ[e] = 0.f; rS1[e] = 0.f; rSd[e] = 0.f; }
  v8f oacc[4];
  #pragma unroll
  for (int t2 = 0; t2 < 4; ++t2) oacc[t2] = zero8f();

  const int qend = q0 + 15;
  for (int j = 0; j <= qend; j += 32) {
    // ---- S tiles: 16 queries x 32 keys ----
    v8f s[2];
    #pragma unroll
    for (int tn = 0; tn < 2; ++tn) {
      v8f sacc = zero8f();
      #pragma unroll
      for (int c = 0; c < 2; ++c) {
        const float* kp = qkv + (size_t)(brow + j + tn * 16 + nl) * (3 * D_)
                        + D_ + h * DH_ + c * 32 + kb;
        v16h bfr;
        #pragma unroll
        for (int e = 0; e < 16; ++e) bfr[e] = (h16)kp[e];
        sacc = wmma32(aq[c], bfr, sacc);
      }
      s[tn] = sacc;
    }
    // ---- scale, distance penalty, causal mask, online softmax ----
    float vm[2][8], dd[2][8];
    int msk[2][8];
    #pragma unroll
    for (int tn = 0; tn < 2; ++tn) {
      const int col = j + tn * 16 + nl;
      #pragma unroll
      for (int e = 0; e < 8; ++e) {
        const int rq = q0 + rsel + e;
        const float d = fabsf((float)(rq - col));
        const int m = (col > rq) ? 1 : 0;
        const float v = s[tn][e] * sscale - dscale * d;
        vm[tn][e]  = m ? -3.0e38f : v;
        dd[tn][e]  = d;
        msk[tn][e] = m;
      }
    }
    #pragma unroll
    for (int e = 0; e < 8; ++e) {
      float mx = fmaxf(vm[0][e], vm[1][e]);
      #pragma unroll
      for (int o = 8; o; o >>= 1) mx = fmaxf(mx, __shfl_xor(mx, o, 32));
      const float mnew = fmaxf(rmax[e], mx);
      const float r = expf(rmax[e] - mnew);
      const float p0 = msk[0][e] ? 0.f : expf(vm[0][e] - mnew);
      const float p1 = msk[1][e] ? 0.f : expf(vm[1][e] - mnew);
      float z  = p0 + p1;
      float s1 = p0 * (vm[0][e] - mnew) + p1 * (vm[1][e] - mnew);
      float sd = p0 * dd[0][e] + p1 * dd[1][e];
      #pragma unroll
      for (int o = 8; o; o >>= 1) {
        z  += __shfl_xor(z,  o, 32);
        s1 += __shfl_xor(s1, o, 32);
        sd += __shfl_xor(sd, o, 32);
      }
      rS1[e]  = r * (rS1[e] + rZ[e] * (rmax[e] - mnew)) + s1;
      rZ[e]   = r * rZ[e] + z;
      rSd[e]  = r * rSd[e] + sd;
      rmax[e] = mnew;
      #pragma unroll
      for (int t2 = 0; t2 < 4; ++t2) oacc[t2][e] *= r;
      Pt[wave][(rsel + e) * 32 + nl]      = (h16)p0;
      Pt[wave][(rsel + e) * 32 + 16 + nl] = (h16)p1;
    }
    // ---- P as A-fragment (via per-wave LDS transpose) ----
    v16h pa;
    {
      const h16* base = &Pt[wave][nl * 32];
      v8h lo = *(const v8h*)(base + koff);
      v8h hi = *(const v8h*)(base + koff + 8);
      #pragma unroll
      for (int e = 0; e < 8; ++e) { pa[e] = lo[e]; pa[e + 8] = hi[e]; }
    }
    // ---- O += P * V ----
    #pragma unroll
    for (int t2 = 0; t2 < 4; ++t2) {
      v16h bv;
      #pragma unroll
      for (int e2 = 0; e2 < 16; ++e2) {
        const int key = j + kb + e2;
        bv[e2] = (h16)qkv[(size_t)(brow + key) * (3 * D_) + 2 * D_ + h * DH_ + t2 * 16 + nl];
      }
      oacc[t2] = wmma32(pa, bv, oacc[t2]);
    }
  }

  // ---- finalize: normalize O, reduce de/fe ----
  #pragma unroll
  for (int e = 0; e < 8; ++e) {
    const float inv = 1.0f / rZ[e];
    const int row = brow + q0 + rsel + e;
    #pragma unroll
    for (int t2 = 0; t2 < 4; ++t2)
      O[(size_t)row * D_ + h * DH_ + t2 * 16 + nl] = oacc[t2][e] * inv;
  }
  if (nl == 0) {
    const float cinv = 1.0f / (float)(B_ * H_ * T_);
    float de = 0.f, fe = 0.f;
    #pragma unroll
    for (int e = 0; e < 8; ++e) {
      de += rSd[e] / rZ[e];
      fe += logf(rZ[e]) - rS1[e] / rZ[e];
    }
    atomicAdd(&acc_defe[0], de * cinv);
    atomicAdd(&acc_defe[1], fe * cinv);
  }
}

// ---------------------------------------------------------------------------
// Per-row cross entropy via online log-sum-exp over V
// ---------------------------------------------------------------------------
__global__ __launch_bounds__(256) void ce_kernel(
    const float* __restrict__ logits, const int* __restrict__ targets,
    float* __restrict__ ce_rows)
{
  const int row = blockIdx.x;
  const float* lr = logits + (size_t)row * V_;
  float m = -3.0e38f, s = 0.f;
  for (int v = threadIdx.x; v < V_; v += 256) {
    const float x = lr[v];
    const float nm = fmaxf(m, x);
    s = s * expf(m - nm) + expf(x - nm);
    m = nm;
  }
  #pragma unroll
  for (int o = 16; o; o >>= 1) {
    const float m2 = __shfl_xor(m, o, 32);
    const float s2 = __shfl_xor(s, o, 32);
    const float nm = fmaxf(m, m2);
    s = s * expf(m - nm) + s2 * expf(m2 - nm);
    m = nm;
  }
  __shared__ float sm[8], ss[8];
  const int wave = threadIdx.x >> 5, lane = threadIdx.x & 31;
  if (lane == 0) { sm[wave] = m; ss[wave] = s; }
  __syncthreads();
  if (threadIdx.x == 0) {
    float M0 = sm[0], S0 = ss[0];
    for (int w = 1; w < 8; ++w) {
      const float nm = fmaxf(M0, sm[w]);
      S0 = S0 * expf(M0 - nm) + ss[w] * expf(sm[w] - nm);
      M0 = nm;
    }
    const float lp = lr[targets[row]] - M0 - logf(S0);
    ce_rows[row] = -lp;
  }
}

__global__ __launch_bounds__(256) void finalize_kernel(
    const float* __restrict__ ce_rows, const float* __restrict__ acc_defe,
    float* __restrict__ out_tail)
{
  float sum = 0.f;
  for (int i = threadIdx.x; i < M_; i += 256) sum += ce_rows[i];
  #pragma unroll
  for (int o = 16; o; o >>= 1) sum += __shfl_xor(sum, o, 32);
  __shared__ float sh[8];
  const int wave = threadIdx.x >> 5, lane = threadIdx.x & 31;
  if (lane == 0) sh[wave] = sum;
  __syncthreads();
  if (threadIdx.x == 0) {
    float t = 0.f;
    for (int w = 0; w < 8; ++w) t += sh[w];
    const float ce = t / (float)M_;
    const float de = acc_defe[0], fe = acc_defe[1];
    out_tail[0] = ce + 0.01f * de + 0.01f * fe;
    out_tail[1] = de;
    out_tail[2] = fe;
  }
}

// ---------------------------------------------------------------------------
// Host orchestration
// ---------------------------------------------------------------------------
extern "C" void kernel_launch(void* const* d_in, const int* in_sizes, int n_in,
                              void* d_out, int out_size, void* d_ws, size_t ws_size,
                              hipStream_t stream)
{
  const int*   idx     = (const int*)d_in[0];
  const int*   targets = (const int*)d_in[1];
  const float* tok_emb = (const float*)d_in[2];
  const float* pos_emb = (const float*)d_in[3];
  const float* qkv_w   = (const float*)d_in[4];
  const float* proj_w  = (const float*)d_in[5];
  const float* dsc     = (const float*)d_in[6];
  const float* ln1_s   = (const float*)d_in[7];
  const float* ln1_b   = (const float*)d_in[8];
  const float* ln2_s   = (const float*)d_in[9];
  const float* ln2_b   = (const float*)d_in[10];
  const float* ffn_w1  = (const float*)d_in[11];
  const float* ffn_b1  = (const float*)d_in[12];
  const float* ffn_w2  = (const float*)d_in[13];
  const float* ffn_b2  = (const float*)d_in[14];
  const float* lnf_s   = (const float*)d_in[15];
  const float* lnf_b   = (const float*)d_in[16];
  float* out = (float*)d_out;
  (void)in_sizes; (void)n_in; (void)out_size; (void)ws_size;

  char* ws = (char*)d_ws;
  size_t off = 0;
  auto alloc = [&](size_t bytes) {
    void* p = ws + off;
    off += (bytes + 255) & ~(size_t)255;
    return p;
  };
  float* X       = (float*)alloc((size_t)M_ * D_ * 4);
  h16*   A16     = (h16*)  alloc((size_t)M_ * DFF_ * 2);
  h16*   W16     = (h16*)  alloc((size_t)V_ * D_ * 2);
  float* S1      = (float*)alloc((size_t)M_ * DFF_ * 4);  // also holds qkv [M,3D]
  float* Obuf    = (float*)alloc((size_t)M_ * D_ * 4);
  float* ce_rows = (float*)alloc((size_t)M_ * 4);
  float* acc     = (float*)alloc(2 * 4);

  (void)hipMemsetAsync(acc, 0, 2 * sizeof(float), stream);

  embed_kernel<<<M_, 256, 0, stream>>>(idx, tok_emb, pos_emb, X);

  for (int l = 0; l < L_; ++l) {
    // ---- attention block ----
    layernorm_f16_kernel<<<M_, 256, 0, stream>>>(X, ln1_s + l * D_, ln1_b + l * D_, A16, D_);
    f32_to_f16_kernel<<<2048, 256, 0, stream>>>(qkv_w + (size_t)l * 3 * D_ * D_, W16,
                                                (size_t)3 * D_ * D_);
    dim3 g1((3 * D_ + BN - 1) / BN, (M_ + BM - 1) / BM);
    gemm_f16_kernel<<<g1, 256, 0, stream>>>(A16, W16, S1, nullptr, M_, 3 * D_, D_);

    dim3 ga(T_ / 64, H_, B_);
    attention_kernel<<<ga, 128, 0, stream>>>(S1, Obuf, dsc + l * H_, acc);

    f32_to_f16_kernel<<<2048, 256, 0, stream>>>(Obuf, A16, (size_t)M_ * D_);
    f32_to_f16_kernel<<<2048, 256, 0, stream>>>(proj_w + (size_t)l * D_ * D_, W16,
                                                (size_t)D_ * D_);
    dim3 g2((D_ + BN - 1) / BN, (M_ + BM - 1) / BM);
    gemm_f16_kernel<<<g2, 256, 0, stream>>>(A16, W16, S1, nullptr, M_, D_, D_);
    add_inplace_kernel<<<2048, 256, 0, stream>>>(X, S1, (size_t)M_ * D_);

    // ---- FFN block ----
    layernorm_f16_kernel<<<M_, 256, 0, stream>>>(X, ln2_s + l * D_, ln2_b + l * D_, A16, D_);
    f32_to_f16_kernel<<<4096, 256, 0, stream>>>(ffn_w1 + (size_t)l * DFF_ * D_, W16,
                                                (size_t)DFF_ * D_);
    dim3 g3((DFF_ + BN - 1) / BN, (M_ + BM - 1) / BM);
    gemm_f16_kernel<<<g3, 256, 0, stream>>>(A16, W16, S1, ffn_b1 + l * DFF_, M_, DFF_, D_);
    gelu_f16_kernel<<<4096, 256, 0, stream>>>(S1, A16, (size_t)M_ * DFF_);
    f32_to_f16_kernel<<<4096, 256, 0, stream>>>(ffn_w2 + (size_t)l * D_ * DFF_, W16,
                                                (size_t)D_ * DFF_);
    dim3 g4((D_ + BN - 1) / BN, (M_ + BM - 1) / BM);
    gemm_f16_kernel<<<g4, 256, 0, stream>>>(A16, W16, Obuf, ffn_b2 + l * D_, M_, D_, DFF_);
    add_inplace_kernel<<<2048, 256, 0, stream>>>(X, Obuf, (size_t)M_ * D_);
  }

  // ---- tied LM head ----
  layernorm_f16_kernel<<<M_, 256, 0, stream>>>(X, lnf_s, lnf_b, A16, D_);
  f32_to_f16_kernel<<<8192, 256, 0, stream>>>(tok_emb, W16, (size_t)V_ * D_);
  dim3 g5((V_ + BN - 1) / BN, (M_ + BM - 1) / BM);
  gemm_f16_kernel<<<g5, 256, 0, stream>>>(A16, W16, out, nullptr, M_, V_, D_);

  // ---- loss ----
  ce_kernel<<<M_, 256, 0, stream>>>(out, targets, ce_rows);
  finalize_kernel<<<1, 256, 0, stream>>>(ce_rows, acc, out + (size_t)M_ * V_);
}
